// DualAttention_80290118632138
// MI455X (gfx1250) — matrix operand order repaired and checked by reading, not compiled, not run on hardware
//
#include <hip/hip_runtime.h>
#include <hip/hip_bf16.h>
#include <math.h>

typedef __attribute__((ext_vector_type(16))) _Float16 v16h;
typedef __attribute__((ext_vector_type(8)))  _Float16 v8h;
typedef __attribute__((ext_vector_type(8)))  float    v8f;

#define DEV __device__ __forceinline__

// ---------------------------------------------------------------------------
// WMMA fragment helpers (CDNA5 ISA 7.12.2 layouts, wave32)
// ---------------------------------------------------------------------------
DEV v16h make_v16(v8h lo, v8h hi) {
  v16h r;
#pragma unroll
  for (int i = 0; i < 8; ++i) { r[i] = lo[i]; r[i + 8] = hi[i]; }
  return r;
}

// A fragment, 16x32 f16 (MxK). lane%16 = row; K elements:
//  e[0..7]  = k0 + (lane/16)*8 + 0..7
//  e[8..15] = k0 + 16 + (lane/16)*8 + 0..7
DEV v16h load_a_frag(const _Float16* __restrict__ A, int lda, int m0, int k0) {
  const int lane = threadIdx.x & 31;
  const _Float16* base =
      A + (size_t)(m0 + (lane & 15)) * lda + k0 + ((lane >> 4) << 3);
  v8h lo = *(const v8h*)(base);
  v8h hi = *(const v8h*)(base + 16);
  return make_v16(lo, hi);
}

// B fragment, 32x16 f16 (KxN), read from a row-major (N,K) "Bt" view.
// lane%16 = column n; K elements: e[0..7] = k0 + (lane/16)*8 + 0..7,
// e[8..15] = k0 + 16 + (lane/16)*8 + 0..7.
DEV v16h load_b_frag(const _Float16* __restrict__ Bt, int ldb, int n0, int k0) {
  const int lane = threadIdx.x & 31;
  const _Float16* base =
      Bt + (size_t)(n0 + (lane & 15)) * ldb + k0 + ((lane >> 4) << 3);
  v8h lo = *(const v8h*)(base);
  v8h hi = *(const v8h*)(base + 16);
  return make_v16(lo, hi);
}

DEV v8f wmma_f16(v16h a, v16h b, v8f c) {
  // (neg_a, A, neg_b, B, c_mod, C, reuse_a, reuse_b)
  return __builtin_amdgcn_wmma_f32_16x16x32_f16(false, a, false, b, (short)0,
                                                c, false, false);
}

// reductions across the 16-lane half (lanes with identical lane>>4)
DEV float rowmax16(float v) {
#pragma unroll
  for (int m = 1; m < 16; m <<= 1) v = fmaxf(v, __shfl_xor(v, m, 32));
  return v;
}
DEV float rowsum16(float v) {
#pragma unroll
  for (int m = 1; m < 16; m <<= 1) v += __shfl_xor(v, m, 32);
  return v;
}

// async global -> LDS copy of one 16-byte chunk (CDNA5, ASYNCcnt tracked)
DEV void async_copy_b128(unsigned lds_byte_off, const _Float16* gsrc) {
  asm volatile("global_load_async_to_lds_b128 %0, %1, off"
               :: "v"(lds_byte_off), "v"(gsrc) : "memory");
}
DEV void wait_async0() {
  asm volatile("s_wait_asynccnt 0x0" ::: "memory");
}

// ---------------------------------------------------------------------------
// Elementwise prep kernels
// ---------------------------------------------------------------------------
__global__ void cast_f32_f16(const float* __restrict__ in,
                             _Float16* __restrict__ out, int n) {
  int i = blockIdx.x * blockDim.x + threadIdx.x;
  if (i < n) out[i] = (_Float16)in[i];
}

// W (K,N) row-major f32  ->  Wt (N,K) row-major f16
__global__ void transpose_cast_w(const float* __restrict__ W,
                                 _Float16* __restrict__ Wt, int K, int N) {
  int i = blockIdx.x * blockDim.x + threadIdx.x;
  if (i >= N * K) return;
  int n = i / K, k = i - n * K;
  Wt[(size_t)n * K + k] = (_Float16)W[(size_t)k * N + n];
}

// (B,S,H,64) f16 -> (B,H,64,S) f16   (per-head transpose for PV fragments)
__global__ void transpose_head(const _Float16* __restrict__ in,
                               _Float16* __restrict__ out) {
  int i = blockIdx.x * blockDim.x + threadIdx.x;  // over 4*1024*8*64
  if (i >= 4 * 1024 * 8 * 64) return;
  int d = i & 63;
  int h = (i >> 6) & 7;
  int s = (i >> 9) & 1023;
  int b = i >> 19;
  out[((((size_t)b * 8 + h) * 64 + d) << 10) + s] = in[i];
}

// RoPE in-place on (B,S,8,64) f16; cos/sin are (S,32) f32
__global__ void rope_f16(_Float16* __restrict__ t,
                         const float* __restrict__ cosb,
                         const float* __restrict__ sinb) {
  int i = blockIdx.x * blockDim.x + threadIdx.x;  // over 4*1024*8*32 pairs
  if (i >= 4 * 1024 * 8 * 32) return;
  int p = i & 31;
  int h = (i >> 5) & 7;
  int s = (i >> 8) & 1023;
  int b = i >> 18;
  float c = cosb[s * 32 + p];
  float sn = sinb[s * 32 + p];
  _Float16* base = t + ((((size_t)b * 1024 + s) * 8 + h) << 6) + 2 * p;
  float xr = (float)base[0], xi = (float)base[1];
  base[0] = (_Float16)(xr * c - xi * sn);
  base[1] = (_Float16)(xr * sn + xi * c);
}

// ---------------------------------------------------------------------------
// WMMA GEMM: C(M,N) = A(M,K) * Bt(N,K)^T
// Block = 256 threads (8 waves) covering one 16-row strip x 512 columns.
// The A strip (16xK f16, <=32KB) is async-staged into LDS once per block;
// each wave then computes four 16x16 tiles, reusing one LDS A-fragment per
// k-step for 4 WMMAs (B panel stays resident in the 192MB L2).
// ---------------------------------------------------------------------------
__global__ __launch_bounds__(256) void wmma_gemm_f16out(
    const _Float16* __restrict__ A, const _Float16* __restrict__ Bt,
    _Float16* __restrict__ C, int N, int K) {
  __shared__ _Float16 atile[16 * 1024];
  const int tid = threadIdx.x;
  const int wave = tid >> 5;
  const int lane = tid & 31;
  const int m0 = blockIdx.y * 16;

  // cooperative async stage: 16 rows x K halves, 16B chunks
  const int strip_bytes = 16 * K * 2;
  for (int bo = tid * 16; bo < strip_bytes; bo += 256 * 16) {
    const int row = bo / (K * 2);
    const int colh = (bo - row * (K * 2)) >> 1;  // halves within row
    async_copy_b128((unsigned)(uintptr_t)&atile[row * K + colh],
                    A + (size_t)(m0 + row) * K + colh);
  }
  wait_async0();
  __syncthreads();

  const int n0 = (blockIdx.x * 8 + wave) * 64;
  if (n0 >= N) return;
  v8f a0 = {}, a1 = {}, a2 = {}, a3 = {};
  for (int k0 = 0; k0 < K; k0 += 32) {
    v16h a = load_a_frag(atile, K, 0, k0);  // ds_load_b128 x2
    a0 = wmma_f16(a, load_b_frag(Bt, K, n0, k0), a0);
    a1 = wmma_f16(a, load_b_frag(Bt, K, n0 + 16, k0), a1);
    a2 = wmma_f16(a, load_b_frag(Bt, K, n0 + 32, k0), a2);
    a3 = wmma_f16(a, load_b_frag(Bt, K, n0 + 48, k0), a3);
  }
  const int col = lane & 15;
#pragma unroll
  for (int r = 0; r < 8; ++r) {
    const size_t ro = (size_t)(m0 + r + 8 * (lane >> 4)) * N + n0;
    C[ro + col] = (_Float16)a0[r];
    C[ro + 16 + col] = (_Float16)a1[r];
    C[ro + 32 + col] = (_Float16)a2[r];
    C[ro + 48 + col] = (_Float16)a3[r];
  }
}

__global__ __launch_bounds__(256) void wmma_gemm_f32out(
    const _Float16* __restrict__ A, const _Float16* __restrict__ Bt,
    float* __restrict__ D, int N, int K, int ldD, int ncol_off) {
  __shared__ _Float16 atile[16 * 1024];
  const int tid = threadIdx.x;
  const int wave = tid >> 5;
  const int lane = tid & 31;
  const int m0 = blockIdx.y * 16;

  const int strip_bytes = 16 * K * 2;
  for (int bo = tid * 16; bo < strip_bytes; bo += 256 * 16) {
    const int row = bo / (K * 2);
    const int colh = (bo - row * (K * 2)) >> 1;
    async_copy_b128((unsigned)(uintptr_t)&atile[row * K + colh],
                    A + (size_t)(m0 + row) * K + colh);
  }
  wait_async0();
  __syncthreads();

  const int n0 = (blockIdx.x * 8 + wave) * 64;
  if (n0 >= N) return;
  v8f a0 = {}, a1 = {}, a2 = {}, a3 = {};
  for (int k0 = 0; k0 < K; k0 += 32) {
    v16h a = load_a_frag(atile, K, 0, k0);
    a0 = wmma_f16(a, load_b_frag(Bt, K, n0, k0), a0);
    a1 = wmma_f16(a, load_b_frag(Bt, K, n0 + 16, k0), a1);
    a2 = wmma_f16(a, load_b_frag(Bt, K, n0 + 32, k0), a2);
    a3 = wmma_f16(a, load_b_frag(Bt, K, n0 + 48, k0), a3);
  }
  const int col = lane & 15;
#pragma unroll
  for (int r = 0; r < 8; ++r) {
    const size_t ro =
        (size_t)(m0 + r + 8 * (lane >> 4)) * ldD + ncol_off + n0;
    D[ro + col] = a0[r];
    D[ro + 16 + col] = a1[r];
    D[ro + 32 + col] = a2[r];
    D[ro + 48 + col] = a3[r];
  }
}

// ---------------------------------------------------------------------------
// Self-attention: flash-style, one wave per (b, head, 16-row tile), causal.
// q,k: (B,S,8,64) f16 (RoPE'd). vT: (B,8,64,S) f16. o: (B,S,512) f16.
// ---------------------------------------------------------------------------
__global__ __launch_bounds__(32) void sa_flash(
    const _Float16* __restrict__ q, const _Float16* __restrict__ k,
    const _Float16* __restrict__ vT, _Float16* __restrict__ o, float scale) {
  __shared__ _Float16 pbuf[16 * 32];
  const int lane = threadIdx.x;
  const int idx = blockIdx.x;
  const int it = idx & 63;
  const int h = (idx >> 6) & 7;
  const int b = idx >> 9;
  const int i0 = it * 16;
  const int cbase = lane & 15;
  const int half = lane >> 4;

  const _Float16* qh = q + (size_t)b * 1024 * 512 + h * 64;
  const _Float16* kh = k + (size_t)b * 1024 * 512 + h * 64;
  const _Float16* vh = vT + ((size_t)(b * 8 + h) * 64) * 1024;

  const v16h qa0 = load_a_frag(qh, 512, i0, 0);
  const v16h qa1 = load_a_frag(qh, 512, i0, 32);

  v8f O0 = {}, O1 = {}, O2 = {}, O3 = {};
  float mrow[8], lrow[8];
#pragma unroll
  for (int r = 0; r < 8; ++r) { mrow[r] = -1e30f; lrow[r] = 0.f; }

  for (int j0 = 0; j0 < i0 + 16; j0 += 32) {
    if (j0 + 32 < i0 + 16) {  // prefetch next K/V tiles (global_prefetch_b8)
      __builtin_prefetch(kh + (size_t)(j0 + 32 + cbase) * 512, 0, 0);
      __builtin_prefetch(vh + (size_t)cbase * 1024 + j0 + 32, 0, 0);
    }
    v8f s0 = {}, s1 = {};
    s0 = wmma_f16(qa0, load_b_frag(kh, 512, j0, 0), s0);
    s0 = wmma_f16(qa1, load_b_frag(kh, 512, j0, 32), s0);
    s1 = wmma_f16(qa0, load_b_frag(kh, 512, j0 + 16, 0), s1);
    s1 = wmma_f16(qa1, load_b_frag(kh, 512, j0 + 16, 32), s1);

    float p0[8], p1[8], alpha[8];
#pragma unroll
    for (int r = 0; r < 8; ++r) {
      const int row = i0 + r + 8 * half;
      float a = s0[r] * scale;
      float c = s1[r] * scale;
      if (j0 + cbase > row) a = -1e30f;
      if (j0 + 16 + cbase > row) c = -1e30f;
      float mnew = fmaxf(mrow[r], rowmax16(fmaxf(a, c)));
      alpha[r] = __expf(mrow[r] - mnew);
      mrow[r] = mnew;
      p0[r] = __expf(a - mnew);
      p1[r] = __expf(c - mnew);
      lrow[r] = lrow[r] * alpha[r] + p0[r] + p1[r];  // per-lane partial
    }
#pragma unroll
    for (int r = 0; r < 8; ++r) {
      O0[r] *= alpha[r]; O1[r] *= alpha[r];
      O2[r] *= alpha[r]; O3[r] *= alpha[r];
    }
    // bounce P through LDS to re-enter WMMA as an A fragment
#pragma unroll
    for (int r = 0; r < 8; ++r) {
      pbuf[(r + 8 * half) * 32 + cbase] = (_Float16)p0[r];
      pbuf[(r + 8 * half) * 32 + 16 + cbase] = (_Float16)p1[r];
    }
    __asm__ volatile("s_wait_dscnt 0x0" ::: "memory");
    v16h ap = load_a_frag(pbuf, 32, 0, 0);
    O0 = wmma_f16(ap, load_b_frag(vh, 1024, 0, j0), O0);
    O1 = wmma_f16(ap, load_b_frag(vh, 1024, 16, j0), O1);
    O2 = wmma_f16(ap, load_b_frag(vh, 1024, 32, j0), O2);
    O3 = wmma_f16(ap, load_b_frag(vh, 1024, 48, j0), O3);
  }

#pragma unroll
  for (int r = 0; r < 8; ++r) {
    const float inv = 1.f / rowsum16(lrow[r]);
    const int row = i0 + r + 8 * half;
    _Float16* orow = o + ((size_t)b * 1024 + row) * 512 + h * 64;
    orow[cbase] = (_Float16)(O0[r] * inv);
    orow[16 + cbase] = (_Float16)(O1[r] * inv);
    orow[32 + cbase] = (_Float16)(O2[r] * inv);
    orow[48 + cbase] = (_Float16)(O3[r] * inv);
  }
}

// ---------------------------------------------------------------------------
// Relational attention: flash softmax over (qa,ka), value = sv (symbols),
// plus rel = tanh(qr.kr^T * rel_scale) recomputed per tile (never
// materialized: avoids 128 MB of HBM traffic), rel-message accumulated with
// the same online-softmax rescaling, wr projection fused into the epilogue.
// ---------------------------------------------------------------------------
__global__ __launch_bounds__(32) void ra_flash(
    const _Float16* __restrict__ qa, const _Float16* __restrict__ ka,
    const _Float16* __restrict__ svT, const _Float16* __restrict__ qr,
    const _Float16* __restrict__ kr, const float* __restrict__ wr,
    _Float16* __restrict__ o, float scale, float rel_scale) {
  __shared__ _Float16 pbuf[16 * 32];
  const int lane = threadIdx.x;
  const int idx = blockIdx.x;
  const int it = idx & 63;
  const int h = (idx >> 6) & 7;
  const int b = idx >> 9;
  const int i0 = it * 16;
  const int cbase = lane & 15;
  const int half = lane >> 4;

  const _Float16* qh = qa + (size_t)b * 1024 * 512 + h * 64;
  const _Float16* kh = ka + (size_t)b * 1024 * 512 + h * 64;
  const _Float16* vh = svT + ((size_t)(b * 8 + h) * 64) * 1024;

  const v16h a0 = load_a_frag(qh, 512, i0, 0);
  const v16h a1 = load_a_frag(qh, 512, i0, 32);

  v8f O0 = {}, O1 = {}, O2 = {}, O3 = {};
  float mrow[8], lrow[8], relacc[8][8];
#pragma unroll
  for (int r = 0; r < 8; ++r) {
    mrow[r] = -1e30f; lrow[r] = 0.f;
#pragma unroll
    for (int g = 0; g < 8; ++g) relacc[g][r] = 0.f;
  }

  for (int j0 = 0; j0 < i0 + 16; j0 += 32) {
    if (j0 + 32 < i0 + 16) {
      __builtin_prefetch(kh + (size_t)(j0 + 32 + cbase) * 512, 0, 0);
      __builtin_prefetch(kr + (size_t)b * 1024 * 512 +
                             (size_t)(j0 + 32 + cbase) * 512, 0, 0);
    }
    v8f s0 = {}, s1 = {};
    s0 = wmma_f16(a0, load_b_frag(kh, 512, j0, 0), s0);
    s0 = wmma_f16(a1, load_b_frag(kh, 512, j0, 32), s0);
    s1 = wmma_f16(a0, load_b_frag(kh, 512, j0 + 16, 0), s1);
    s1 = wmma_f16(a1, load_b_frag(kh, 512, j0 + 16, 32), s1);

    float p0[8], p1[8], alpha[8];
#pragma unroll
    for (int r = 0; r < 8; ++r) {
      const int row = i0 + r + 8 * half;
      float a = s0[r] * scale;
      float c = s1[r] * scale;
      if (j0 + cbase > row) a = -1e30f;
      if (j0 + 16 + cbase > row) c = -1e30f;
      float mnew = fmaxf(mrow[r], rowmax16(fmaxf(a, c)));
      alpha[r] = __expf(mrow[r] - mnew);
      mrow[r] = mnew;
      p0[r] = __expf(a - mnew);
      p1[r] = __expf(c - mnew);
      lrow[r] = lrow[r] * alpha[r] + p0[r] + p1[r];
    }
#pragma unroll
    for (int r = 0; r < 8; ++r) {
      O0[r] *= alpha[r]; O1[r] *= alpha[r];
      O2[r] *= alpha[r]; O3[r] *= alpha[r];
    }
#pragma unroll
    for (int r = 0; r < 8; ++r) {
      pbuf[(r + 8 * half) * 32 + cbase] = (_Float16)p0[r];
      pbuf[(r + 8 * half) * 32 + 16 + cbase] = (_Float16)p1[r];
    }
    __asm__ volatile("s_wait_dscnt 0x0" ::: "memory");
    v16h ap = load_a_frag(pbuf, 32, 0, 0);
    O0 = wmma_f16(ap, load_b_frag(vh, 1024, 0, j0), O0);
    O1 = wmma_f16(ap, load_b_frag(vh, 1024, 16, j0), O1);
    O2 = wmma_f16(ap, load_b_frag(vh, 1024, 32, j0), O2);
    O3 = wmma_f16(ap, load_b_frag(vh, 1024, 48, j0), O3);

    // rel = tanh(qr.kr^T * rel_scale) per rel-head; rel_msg += P .* rel
#pragma unroll
    for (int g = 0; g < 8; ++g) {
      const _Float16* qrg = qr + (size_t)b * 1024 * 512 + g * 64;
      const _Float16* krg = kr + (size_t)b * 1024 * 512 + g * 64;
      v16h g0 = load_a_frag(qrg, 512, i0, 0);
      v16h g1 = load_a_frag(qrg, 512, i0, 32);
      v8f r0 = {}, r1 = {};
      r0 = wmma_f16(g0, load_b_frag(krg, 512, j0, 0), r0);
      r0 = wmma_f16(g1, load_b_frag(krg, 512, j0, 32), r0);
      r1 = wmma_f16(g0, load_b_frag(krg, 512, j0 + 16, 0), r1);
      r1 = wmma_f16(g1, load_b_frag(krg, 512, j0 + 16, 32), r1);
#pragma unroll
      for (int r = 0; r < 8; ++r) {
        float t0 = tanhf(r0[r] * rel_scale);
        float t1 = tanhf(r1[r] * rel_scale);
        relacc[g][r] = relacc[g][r] * alpha[r] + p0[r] * t0 + p1[r] * t1;
      }
    }
  }

#pragma unroll
  for (int r = 0; r < 8; ++r) {
    const float inv = 1.f / rowsum16(lrow[r]);
    float rm[8];
#pragma unroll
    for (int g = 0; g < 8; ++g) rm[g] = rowsum16(relacc[g][r]) * inv;
    const int row = i0 + r + 8 * half;
    float od[4] = {O0[r] * inv, O1[r] * inv, O2[r] * inv, O3[r] * inv};
#pragma unroll
    for (int c = 0; c < 4; ++c) {
      const int d = c * 16 + cbase;
      const float* wrp = wr + ((size_t)h * 64 + d) * 8;  // wr (8,64,8)
      float acc = od[c];
#pragma unroll
      for (int g = 0; g < 8; ++g) acc += rm[g] * wrp[g];
      o[((size_t)b * 1024 + row) * 512 + h * 64 + d] = (_Float16)acc;
    }
  }
}

// ---------------------------------------------------------------------------
// Launch
// ---------------------------------------------------------------------------
extern "C" void kernel_launch(void* const* d_in, const int* in_sizes, int n_in,
                              void* d_out, int out_size, void* d_ws,
                              size_t ws_size, hipStream_t stream) {
  (void)in_sizes; (void)n_in; (void)out_size; (void)ws_size;
  const float* x = (const float*)d_in[0];
  const float* symbols = (const float*)d_in[1];
  const float* fcos = (const float*)d_in[2];
  const float* fsin = (const float*)d_in[3];
  const float* w_proj[8] = {(const float*)d_in[4],  (const float*)d_in[5],
                            (const float*)d_in[6],  (const float*)d_in[8],
                            (const float*)d_in[9],  (const float*)d_in[10],
                            (const float*)d_in[11], (const float*)d_in[13]};
  const float* wo_sa = (const float*)d_in[7];
  const float* wr = (const float*)d_in[12];
  const float* wo_ra = (const float*)d_in[14];
  float* out = (float*)d_out;

  const int S = 1024, D = 1024, P = 512;
  const size_t BS = 4096;

  char* ws = (char*)d_ws;
  size_t off = 0;
  auto alloc = [&](size_t bytes) -> _Float16* {
    _Float16* p = (_Float16*)(ws + off);
    off += (bytes + 255) & ~(size_t)255;
    return p;
  };

  _Float16* x16 = alloc(BS * D * 2);
  _Float16* sym16 = alloc(BS * D * 2);
  _Float16* wt[8];
  for (int i = 0; i < 8; ++i) wt[i] = alloc((size_t)P * D * 2);
  _Float16* wo_sa_t = alloc((size_t)P * P * 2);
  _Float16* wo_ra_t = alloc((size_t)P * P * 2);
  _Float16* qsa = alloc(BS * P * 2);
  _Float16* ksa = alloc(BS * P * 2);
  _Float16* vsa = alloc(BS * P * 2);
  _Float16* qat = alloc(BS * P * 2);
  _Float16* kat = alloc(BS * P * 2);
  _Float16* qrl = alloc(BS * P * 2);
  _Float16* krl = alloc(BS * P * 2);
  _Float16* svv = alloc(BS * P * 2);
  _Float16* vT = alloc(BS * P * 2);
  _Float16* svT = alloc(BS * P * 2);
  _Float16* sacat = alloc(BS * P * 2);
  _Float16* racat = alloc(BS * P * 2);

  const int nXD = (int)(BS * D);
  cast_f32_f16<<<(nXD + 255) / 256, 256, 0, stream>>>(x, x16, nXD);
  cast_f32_f16<<<(nXD + 255) / 256, 256, 0, stream>>>(symbols, sym16, nXD);
  for (int i = 0; i < 8; ++i)
    transpose_cast_w<<<(D * P + 255) / 256, 256, 0, stream>>>(w_proj[i], wt[i],
                                                              D, P);
  transpose_cast_w<<<(P * P + 255) / 256, 256, 0, stream>>>(wo_sa, wo_sa_t, P,
                                                            P);
  transpose_cast_w<<<(P * P + 255) / 256, 256, 0, stream>>>(wo_ra, wo_ra_t, P,
                                                            P);

  // projections: (4096x1024) x (1024x512); one block per 16-row strip
  dim3 gg(P / 512, (unsigned)(BS / 16));
  wmma_gemm_f16out<<<gg, 256, 0, stream>>>(x16, wt[0], qsa, P, D);
  wmma_gemm_f16out<<<gg, 256, 0, stream>>>(x16, wt[1], ksa, P, D);
  wmma_gemm_f16out<<<gg, 256, 0, stream>>>(x16, wt[2], vsa, P, D);
  wmma_gemm_f16out<<<gg, 256, 0, stream>>>(x16, wt[3], qat, P, D);
  wmma_gemm_f16out<<<gg, 256, 0, stream>>>(x16, wt[4], kat, P, D);
  wmma_gemm_f16out<<<gg, 256, 0, stream>>>(x16, wt[5], qrl, P, D);
  wmma_gemm_f16out<<<gg, 256, 0, stream>>>(x16, wt[6], krl, P, D);
  wmma_gemm_f16out<<<gg, 256, 0, stream>>>(sym16, wt[7], svv, P, D);

  const int nRope = 4 * 1024 * 8 * 32;
  rope_f16<<<(nRope + 255) / 256, 256, 0, stream>>>(qsa, fcos, fsin);
  rope_f16<<<(nRope + 255) / 256, 256, 0, stream>>>(ksa, fcos, fsin);
  rope_f16<<<(nRope + 255) / 256, 256, 0, stream>>>(qat, fcos, fsin);
  rope_f16<<<(nRope + 255) / 256, 256, 0, stream>>>(kat, fcos, fsin);

  const int nTr = 4 * 1024 * 8 * 64;
  transpose_head<<<(nTr + 255) / 256, 256, 0, stream>>>(vsa, vT);
  transpose_head<<<(nTr + 255) / 256, 256, 0, stream>>>(svv, svT);

  const float scale = 0.125f;      // 1/sqrt(64)
  const float rel_scale = 0.125f;  // 1/sqrt(64)
  sa_flash<<<4 * 8 * (S / 16), 32, 0, stream>>>(qsa, ksa, vT, sacat, scale);
  ra_flash<<<4 * 8 * (S / 16), 32, 0, stream>>>(qat, kat, svT, qrl, krl, wr,
                                                racat, scale, rel_scale);

  // final projections, written straight into d_out halves (f32)
  wmma_gemm_f32out<<<gg, 256, 0, stream>>>(sacat, wo_sa_t, out, P, P, 1024, 0);
  wmma_gemm_f32out<<<gg, 256, 0, stream>>>(racat, wo_ra_t, out, P, P, 1024,
                                           512);
}
